// TopKSAE_10256381902965
// MI455X (gfx1250) — compile-verified, matrix operand order checked by reading, not run on hardware
//
#include <hip/hip_runtime.h>
#include <hip/hip_bf16.h>

#define N_TOKENS   8192
#define INPUT_DIM  1024
#define HIDDEN_DIM 16384
#define TOPK       64

typedef __attribute__((ext_vector_type(16))) __bf16 v16bf;
typedef __attribute__((ext_vector_type(8)))  float  v8f;

// Native clang vector int4 (NOT HIP's HIP_vector_type<int,4>), matching the
// async-to-LDS builtin parameter: 'vector_size(16) int __device__(AS1) *'.
typedef int vint4 __attribute__((__vector_size__(4 * sizeof(int))));
typedef __attribute__((address_space(1))) vint4 as1_vint4;
typedef __attribute__((address_space(3))) vint4 as3_vint4;

#if __has_builtin(__builtin_amdgcn_global_load_async_to_lds_b128)
#define USE_ASYNC_LDS 1
#else
#define USE_ASYNC_LDS 0
#endif

#if __has_builtin(__builtin_amdgcn_s_wait_asynccnt)
#define WAIT_ASYNC(n) __builtin_amdgcn_s_wait_asynccnt(n)
#else
#define WAIT_ASYNC(n) asm volatile("s_wait_asynccnt %0" ::"i"(n))
#endif

union Frag32 {
    uint4 u[2];
    v16bf v;
};

// Fragment layout per CDNA5 ISA 16-bit A-matrix table (7.12.2):
//   lanes 0-15 : K = 0..7 in elems 0..7, K = 16..23 in elems 8..15
//   lanes 16-31: K = 8..15 / 24..31
// LDS tile row-major [row][32 K halfs] -> two 16B LDS reads per fragment.
__device__ __forceinline__ v16bf load_frag(const __bf16* __restrict__ base, int row, int kb) {
    Frag32 f;
    const __bf16* p = base + row * 32;
    f.u[0] = *(const uint4*)(p + kb);
    f.u[1] = *(const uint4*)(p + kb + 16);
    return f.v;
}

// Copy 16 consecutive bf16 (32B) global->LDS: async DMA if available, else regs.
__device__ __forceinline__ void stage32(const __bf16* g, __bf16* l) {
#if USE_ASYNC_LDS
    __builtin_amdgcn_global_load_async_to_lds_b128((as1_vint4*)g, (as3_vint4*)l, 0, 0);
    __builtin_amdgcn_global_load_async_to_lds_b128((as1_vint4*)(g + 8), (as3_vint4*)(l + 8), 0, 0);
#else
    *(uint4*)l = *(const uint4*)g;
    *(uint4*)(l + 8) = *(const uint4*)(g + 8);
#endif
}

// ---------------------------------------------------------------------------
// Kernel 0: split fp32 -> bf16 hi + bf16 lo (fp32 ~ hi + lo, error ~2^-16 rel)
// ---------------------------------------------------------------------------
__global__ __launch_bounds__(256) void split_bf16(
    const float* __restrict__ src, __bf16* __restrict__ hi, __bf16* __restrict__ lo)
{
    const size_t i = (size_t)blockIdx.x * blockDim.x + threadIdx.x;
    float4 v = ((const float4*)src)[i];
    float a[4] = {v.x, v.y, v.z, v.w};
#pragma unroll
    for (int j = 0; j < 4; ++j) {
        __bf16 h = (__bf16)a[j];
        hi[i * 4 + j] = h;
        lo[i * 4 + j] = (__bf16)(a[j] - (float)h);
    }
}

// ---------------------------------------------------------------------------
// Kernel 1: z = x @ W_enc^T + b_enc, split-bf16 (hi*hi + lo*hi + hi*lo) WMMA.
// Block tile 128x128, 8 waves in 4(M)x2(N); each wave: 2x4 16x16 tiles.
// Double-buffered LDS staged with GLOBAL_LOAD_ASYNC_TO_LDS (ASYNCcnt).
// ---------------------------------------------------------------------------
__global__ __launch_bounds__(256) void sae_encode_wmma(
    const __bf16* __restrict__ xh, const __bf16* __restrict__ xl,
    const __bf16* __restrict__ Wh, const __bf16* __restrict__ Wl,
    const float* __restrict__ benc, float* __restrict__ z)
{
    // [buf][hi/lo][128 rows * 32 K]
    __shared__ alignas(16) __bf16 sA[2][2][128 * 32];
    __shared__ alignas(16) __bf16 sB[2][2][128 * 32];

    const int t    = threadIdx.x;
    const int lane = t & 31;
    const int w    = t >> 5;
    const int wm   = w >> 1;            // 0..3  -> 32-row strip
    const int wn   = w & 1;             // 0..1  -> 64-col strip
    const int row0 = blockIdx.y * 128;  // token rows
    const int col0 = blockIdx.x * 128;  // hidden cols

    v8f acc[2][4];
    const v8f vzero = {0.f, 0.f, 0.f, 0.f, 0.f, 0.f, 0.f, 0.f};
#pragma unroll
    for (int mt = 0; mt < 2; ++mt)
#pragma unroll
        for (int nt = 0; nt < 4; ++nt) acc[mt][nt] = vzero;

    // staging map: 2 threads per row, 32B each
    const int sr = t >> 1;              // 0..127
    const int sc = (t & 1) << 4;        // 0 or 16 halfs

    const int kb = (lane >> 4) << 3;    // 0 or 8
    const int fr = lane & 15;

    auto issue = [&](int b, int k0) {
        const size_t ga = (size_t)(row0 + sr) * INPUT_DIM + k0 + sc;
        const size_t gb = (size_t)(col0 + sr) * INPUT_DIM + k0 + sc;
        const int lo = sr * 32 + sc;
        stage32(xh + ga, &sA[b][0][lo]);
        stage32(xl + ga, &sA[b][1][lo]);
        stage32(Wh + gb, &sB[b][0][lo]);
        stage32(Wl + gb, &sB[b][1][lo]);
    };

    issue(0, 0);   // preload first tile
    int buf = 0;
    for (int k0 = 0; k0 < INPUT_DIM; k0 += 32) {
        const bool more = (k0 + 32 < INPUT_DIM);
        if (more) issue(buf ^ 1, k0 + 32);
#if USE_ASYNC_LDS
        if (more) WAIT_ASYNC(8);   // leave the 8 just-issued in flight
        else      WAIT_ASYNC(0);
#endif
        __syncthreads();           // current buffer visible to all waves

        v16bf ah[2], al[2];
#pragma unroll
        for (int mt = 0; mt < 2; ++mt) {
            const int ar = (wm << 5) + (mt << 4) + fr;
            ah[mt] = load_frag(&sA[buf][0][0], ar, kb);
            al[mt] = load_frag(&sA[buf][1][0], ar, kb);
        }
#pragma unroll
        for (int nt = 0; nt < 4; ++nt) {
            const int br = (wn << 6) + (nt << 4) + fr;
            v16bf bh = load_frag(&sB[buf][0][0], br, kb);
            v16bf bl = load_frag(&sB[buf][1][0], br, kb);
#pragma unroll
            for (int mt = 0; mt < 2; ++mt) {
                acc[mt][nt] = __builtin_amdgcn_wmma_f32_16x16x32_bf16(
                    false, ah[mt], false, bh, (short)0, acc[mt][nt], false, false);
                acc[mt][nt] = __builtin_amdgcn_wmma_f32_16x16x32_bf16(
                    false, al[mt], false, bh, (short)0, acc[mt][nt], false, false);
                acc[mt][nt] = __builtin_amdgcn_wmma_f32_16x16x32_bf16(
                    false, ah[mt], false, bl, (short)0, acc[mt][nt], false, false);
            }
        }
        __syncthreads();           // everyone done reading before overwrite
        buf ^= 1;
    }

    // Epilogue: C/D layout -- VGPR r holds M=r (lanes 0-15) / M=8+r (lanes 16-31)
    const int n_in  = lane & 15;
    const int m_off = (lane >> 4) << 3;
#pragma unroll
    for (int nt = 0; nt < 4; ++nt) {
        const int col = col0 + (wn << 6) + (nt << 4) + n_in;
        const float bias = benc[col];
#pragma unroll
        for (int mt = 0; mt < 2; ++mt) {
#pragma unroll
            for (int r = 0; r < 8; ++r) {
                const int row = row0 + (wm << 5) + (mt << 4) + m_off + r;
                z[(size_t)row * HIDDEN_DIM + col] = acc[mt][nt][r] + bias;
            }
        }
    }
}

// ---------------------------------------------------------------------------
// Kernel 2: exact per-row top-64 via bit-pattern binary search (regs-resident).
// Zeroes non-top-k entries of z in place; emits compact (val, idx) lists.
// ---------------------------------------------------------------------------
__global__ __launch_bounds__(256) void sae_topk(
    float* __restrict__ z, float* __restrict__ cvals, int* __restrict__ cidx)
{
    const int row = blockIdx.x;
    const int t   = threadIdx.x;
    float* zr = z + (size_t)row * HIDDEN_DIM;

    unsigned key[64];
#pragma unroll
    for (int j = 0; j < 64; ++j) {
        unsigned b = __float_as_uint(zr[j * 256 + t]);
        key[j] = (b & 0x80000000u) ? ~b : (b | 0x80000000u);  // order-preserving
    }

    __shared__ int red[256];
    __shared__ int s_eq, s_keep;

    unsigned lo = 0u, hi = 0xFFFFFFFFu;
    while (lo < hi) {
        unsigned mid = (unsigned)(((unsigned long long)lo + (unsigned long long)hi + 1ull) >> 1);
        int c = 0;
#pragma unroll
        for (int j = 0; j < 64; ++j) c += (key[j] >= mid) ? 1 : 0;
        red[t] = c;
        __syncthreads();
        for (int s = 128; s > 0; s >>= 1) {
            if (t < s) red[t] += red[t + s];
            __syncthreads();
        }
        int total = red[0];
        __syncthreads();
        if (total >= TOPK) lo = mid; else hi = mid - 1;
    }
    const unsigned T = lo;

    int cg = 0;
#pragma unroll
    for (int j = 0; j < 64; ++j) cg += (key[j] > T) ? 1 : 0;
    red[t] = cg;
    __syncthreads();
    for (int s = 128; s > 0; s >>= 1) {
        if (t < s) red[t] += red[t + s];
        __syncthreads();
    }
    const int cnt_gt = red[0];
    if (t == 0) { s_eq = 0; s_keep = 0; }
    __syncthreads();
    const int quota = TOPK - cnt_gt;

#pragma unroll
    for (int j = 0; j < 64; ++j) {
        unsigned u = key[j];
        bool keep = false;
        if (u > T) keep = true;
        else if (u == T) { int p = atomicAdd(&s_eq, 1); keep = (p < quota); }
        unsigned b = (u & 0x80000000u) ? (u & 0x7FFFFFFFu) : ~u;
        float f = __uint_as_float(b);
        int col = j * 256 + t;
        zr[col] = keep ? f : 0.0f;
        if (keep) {
            int c = atomicAdd(&s_keep, 1);
            cvals[row * TOPK + c] = f;
            cidx [row * TOPK + c] = col;
        }
    }
}

// ---------------------------------------------------------------------------
// Kernel 3: W_dec [I,H] -> W_decT [H,I] for contiguous-row decode gathers.
// ---------------------------------------------------------------------------
__global__ __launch_bounds__(256) void transpose_wdec(
    const float* __restrict__ Wd, float* __restrict__ WdT)
{
    __shared__ float tile[32][33];
    const int hb = blockIdx.x * 32;
    const int ib = blockIdx.y * 32;
    const int tx = threadIdx.x;
    const int ty = threadIdx.y;
#pragma unroll
    for (int j = 0; j < 4; ++j)
        tile[ty + j * 8][tx] = Wd[(size_t)(ib + ty + j * 8) * HIDDEN_DIM + hb + tx];
    __syncthreads();
#pragma unroll
    for (int j = 0; j < 4; ++j)
        WdT[(size_t)(hb + ty + j * 8) * INPUT_DIM + ib + tx] = tile[tx][ty + j * 8];
}

// ---------------------------------------------------------------------------
// Kernel 4: sparse decode; W_decT (64 MB) is L2-resident (192 MB L2).
// ---------------------------------------------------------------------------
__global__ __launch_bounds__(256) void sae_decode(
    const float* __restrict__ cvals, const int* __restrict__ cidx,
    const float* __restrict__ WdT, const float* __restrict__ bdec,
    float* __restrict__ recon)
{
    const int n = blockIdx.x;
    const int t = threadIdx.x;
    __shared__ float sv[TOPK];
    __shared__ int   si[TOPK];
    if (t < TOPK) { sv[t] = cvals[n * TOPK + t]; si[t] = cidx[n * TOPK + t]; }
    __syncthreads();

    float a0 = 0.f, a1 = 0.f, a2 = 0.f, a3 = 0.f;
#pragma unroll 4
    for (int k = 0; k < TOPK; ++k) {
        float v = sv[k];
        const float* wp = WdT + (size_t)si[k] * INPUT_DIM;
        a0 += v * wp[t];
        a1 += v * wp[t + 256];
        a2 += v * wp[t + 512];
        a3 += v * wp[t + 768];
    }
    float* ro = recon + (size_t)n * INPUT_DIM;
    ro[t]       = a0 + bdec[t];
    ro[t + 256] = a1 + bdec[t + 256];
    ro[t + 512] = a2 + bdec[t + 512];
    ro[t + 768] = a3 + bdec[t + 768];
}

extern "C" void kernel_launch(void* const* d_in, const int* in_sizes, int n_in,
                              void* d_out, int out_size, void* d_ws, size_t ws_size,
                              hipStream_t stream) {
    const float* x    = (const float*)d_in[0];
    const float* Wenc = (const float*)d_in[1];
    const float* benc = (const float*)d_in[2];
    const float* Wdec = (const float*)d_in[3];
    const float* bdec = (const float*)d_in[4];

    float* out   = (float*)d_out;
    float* recon = out;                                 // [N, I]
    float* z     = out + (size_t)N_TOKENS * INPUT_DIM;  // [N, H] (becomes z_sparse)

    // workspace: W_decT 64MB | cvals 2MB | cidx 2MB | xh 16MB | xl 16MB | Wh 32MB | Wl 32MB
    float*  WdT   = (float*)d_ws;
    float*  cvals = WdT + (size_t)HIDDEN_DIM * INPUT_DIM;
    int*    cidx  = (int*)(cvals + (size_t)N_TOKENS * TOPK);
    __bf16* xhp   = (__bf16*)(cidx + (size_t)N_TOKENS * TOPK);
    __bf16* xlp   = xhp + (size_t)N_TOKENS * INPUT_DIM;
    __bf16* Whp   = xlp + (size_t)N_TOKENS * INPUT_DIM;
    __bf16* Wlp   = Whp + (size_t)HIDDEN_DIM * INPUT_DIM;

    split_bf16<<<dim3((N_TOKENS * INPUT_DIM / 4) / 256), dim3(256), 0, stream>>>(x, xhp, xlp);
    split_bf16<<<dim3((HIDDEN_DIM * INPUT_DIM / 4) / 256), dim3(256), 0, stream>>>(Wenc, Whp, Wlp);
    transpose_wdec<<<dim3(HIDDEN_DIM / 32, INPUT_DIM / 32), dim3(32, 8), 0, stream>>>(Wdec, WdT);
    sae_encode_wmma<<<dim3(HIDDEN_DIM / 128, N_TOKENS / 128), dim3(256), 0, stream>>>(
        xhp, xlp, Whp, Wlp, benc, z);
    sae_topk<<<dim3(N_TOKENS), dim3(256), 0, stream>>>(z, cvals, cidx);
    sae_decode<<<dim3(N_TOKENS), dim3(256), 0, stream>>>(cvals, cidx, WdT, bdec, recon);
}
